// GineEncoder_80882824118336
// MI455X (gfx1250) — matrix operand order repaired and checked by reading, not compiled, not run on hardware
//
#include <hip/hip_runtime.h>

typedef __attribute__((ext_vector_type(16))) _Float16 v16h;
typedef __attribute__((ext_vector_type(8)))  float    v8f;

#define LDH 64   // padded channel stride for all node tensors

// ---------------------------------------------------------------- utilities

__global__ void zero_kernel(float* __restrict__ p, int n) {
  int i = blockIdx.x * blockDim.x + threadIdx.x;
  if (i < n) p[i] = 0.0f;
}

// copy x[N,32] into h[N,64] (cols 32..63 zero)
__global__ void init_h_kernel(const float* __restrict__ x, float* __restrict__ h, int nNodes) {
  int i = blockIdx.x * blockDim.x + threadIdx.x;
  if (i < nNodes * LDH) {
    int n = i >> 6, ch = i & 63;
    h[i] = (ch < 32) ? x[n * 32 + ch] : 0.0f;
  }
}

// ------------------------------------------- fused edge projection + scatter
// e = edge_attr @ edge_W^T + edge_b  (WMMA f16->f32, K=16 padded to 32)
// msg = relu(h[src] + e); aggr[dst] += msg  (hw fp32 atomics, L2 resident)
template<int CT>   // out-channel tiles: 2 (in_c=32) or 4 (in_c=64)
__global__ void edge_scatter_kernel(const float* __restrict__ eattr,  // [E,16]
                                    const int*   __restrict__ src,    // [E]
                                    const int*   __restrict__ dst,    // [E]
                                    const float* __restrict__ h,      // [N,64]
                                    const float* __restrict__ eW,     // [CT*16,16]
                                    const float* __restrict__ eB,     // [CT*16]
                                    float* __restrict__ aggr,         // [N,64]
                                    int nEdges)
{
  const int lane   = threadIdx.x & 31;
  const int wid    = (blockIdx.x * blockDim.x + threadIdx.x) >> 5;
  const int nwaves = (gridDim.x * blockDim.x) >> 5;
  const int half   = lane >> 4;      // K-half select (ISA 16-bit A/B layouts)
  const int l16    = lane & 15;

  // B tiles: B[k][n] = edge_W[c*16+n][k], k<16; k>=16 (upper K half) zero.
  // lane = n + 16*b holds halves 0..7 -> k = b*8 + h; halves 8..15 (k>=16) zero
  v16h Bt[CT];
  #pragma unroll
  for (int c = 0; c < CT; ++c) {
    const float4* wrow = (const float4*)(eW + (c * 16 + l16) * 16);
    float4 w0 = wrow[0], w1 = wrow[1], w2 = wrow[2], w3 = wrow[3];
    float wv[16] = {w0.x,w0.y,w0.z,w0.w, w1.x,w1.y,w1.z,w1.w,
                    w2.x,w2.y,w2.z,w2.w, w3.x,w3.y,w3.z,w3.w};
    v16h bv;
    #pragma unroll
    for (int k = 0; k < 16; ++k)
      bv[k] = (k < 8) ? (_Float16)wv[half * 8 + k] : (_Float16)0.0f;
    Bt[c] = bv;
  }
  float biasv[CT];
  #pragma unroll
  for (int c = 0; c < CT; ++c) biasv[c] = eB[c * 16 + l16];

  const int tiles = nEdges >> 4;     // 100k tiles, exact
  for (int t = wid; t < tiles; t += nwaves) {
    const int e0 = t << 4;
    // A: lane holds edge row e0+l16, features half*8..+7 in halves 0..7; K>=16 zero
    const float4* ap = (const float4*)(eattr + (e0 + l16) * 16 + half * 8);
    float4 a0 = ap[0], a1 = ap[1];
    v16h A = {};
    A[0]=(_Float16)a0.x; A[1]=(_Float16)a0.y; A[2]=(_Float16)a0.z; A[3]=(_Float16)a0.w;
    A[4]=(_Float16)a1.x; A[5]=(_Float16)a1.y; A[6]=(_Float16)a1.z; A[7]=(_Float16)a1.w;

    // this lane owns D rows r + 8*half
    const int4* sp = (const int4*)(src + e0 + 8 * half);
    const int4* dp = (const int4*)(dst + e0 + 8 * half);
    int4 s0 = sp[0], s1 = sp[1], d0 = dp[0], d1 = dp[1];
    int s8[8] = {s0.x,s0.y,s0.z,s0.w, s1.x,s1.y,s1.z,s1.w};
    int d8[8] = {d0.x,d0.y,d0.z,d0.w, d1.x,d1.y,d1.z,d1.w};

    #pragma unroll
    for (int c = 0; c < CT; ++c) {
      v8f C;
      #pragma unroll
      for (int r = 0; r < 8; ++r) C[r] = biasv[c];
      C = __builtin_amdgcn_wmma_f32_16x16x32_f16(false, A, false, Bt[c],
                                                 (short)0, C, false, false);
      const int col = c * 16 + l16;
      // batch the 8 gathers so they issue back-to-back under one wait
      float hv[8];
      #pragma unroll
      for (int r = 0; r < 8; ++r)
        hv[r] = h[s8[r] * LDH + col];            // 32-bit index -> saddr form
      #pragma unroll
      for (int r = 0; r < 8; ++r) {
        float m = hv[r] + C[r];
        m = m > 0.0f ? m : 0.0f;
        unsafeAtomicAdd(&aggr[d8[r] * LDH + col], m);
      }
    }
  }
}

// ------------------------------------------------ WMMA linear (+BN stats)
// out[n, 0..63] = act(in)[n, 0..32*QMAX-1] @ W^T + bias
//   MODE 0: act = in0 + in1                      (h + aggr; GINE eps=0)
//   MODE 1: act = relu(sb[ch]*in0 + sb[64+ch])   (apply previous BN+ReLU)
// also accumulates per-channel sum / sumsq of out into stats[0..63]/[64..127]
template<int QMAX, int MODE>
__global__ void linear_wmma_kernel(const float* __restrict__ in0,
                                   const float* __restrict__ in1,
                                   const float* __restrict__ sb,
                                   const float* __restrict__ W,      // [64, 32*QMAX]
                                   const float* __restrict__ bias,   // [64]
                                   float* __restrict__ out,          // [N,64]
                                   float* __restrict__ stats,        // [128]
                                   int nNodes)
{
  const int lane   = threadIdx.x & 31;
  const int wid    = (blockIdx.x * blockDim.x + threadIdx.x) >> 5;
  const int nwaves = (gridDim.x * blockDim.x) >> 5;
  const int half   = lane >> 4, l16 = lane & 15;
  const int dIn    = QMAX * 32;

  // B[c][q][k][n] = W[c*16+n][q*32+k]; lane n+16*b holds k = b*8 + h (h<8), 16 + b*8 + (h-8) (h>=8)
  v16h B[4][QMAX];
  #pragma unroll
  for (int c = 0; c < 4; ++c)
    #pragma unroll
    for (int q = 0; q < QMAX; ++q) {
      const float4* wrow = (const float4*)(W + (c * 16 + l16) * dIn + q * 32 + half * 8);
      float4 w0 = wrow[0], w1 = wrow[1], w2 = wrow[4], w3 = wrow[5];
      float wv[16] = {w0.x,w0.y,w0.z,w0.w, w1.x,w1.y,w1.z,w1.w,
                      w2.x,w2.y,w2.z,w2.w, w3.x,w3.y,w3.z,w3.w};
      v16h bv;
      #pragma unroll
      for (int k = 0; k < 16; ++k) bv[k] = (_Float16)wv[k];
      B[c][q] = bv;
    }
  float bcol[4];
  #pragma unroll
  for (int c = 0; c < 4; ++c) bcol[c] = bias[c * 16 + l16];

  // hoisted per-lane BN scale/bias (MODE 1): halves 0..7 -> ch q*32+half*8+k,
  // halves 8..15 -> +16
  float scv[QMAX][16], biv[QMAX][16];
  if (MODE == 1) {
    #pragma unroll
    for (int q = 0; q < QMAX; ++q) {
      const int b0 = q * 32 + half * 8;
      const float4* s0p = (const float4*)(sb + b0);
      const float4* s1p = (const float4*)(sb + b0 + 16);
      const float4* b0p = (const float4*)(sb + 64 + b0);
      const float4* b1p = (const float4*)(sb + 64 + b0 + 16);
      float4 sA = s0p[0], sB = s0p[1], sC = s1p[0], sD = s1p[1];
      float4 bA = b0p[0], bB = b0p[1], bC = b1p[0], bD = b1p[1];
      float st[16] = {sA.x,sA.y,sA.z,sA.w, sB.x,sB.y,sB.z,sB.w,
                      sC.x,sC.y,sC.z,sC.w, sD.x,sD.y,sD.z,sD.w};
      float bt[16] = {bA.x,bA.y,bA.z,bA.w, bB.x,bB.y,bB.z,bB.w,
                      bC.x,bC.y,bC.z,bC.w, bD.x,bD.y,bD.z,bD.w};
      #pragma unroll
      for (int k = 0; k < 16; ++k) { scv[q][k] = st[k]; biv[q][k] = bt[k]; }
    }
  }

  const int tiles = nNodes >> 4;     // 3125, exact
  float psum[4] = {0, 0, 0, 0}, psq[4] = {0, 0, 0, 0};

  for (int t = wid; t < tiles; t += nwaves) {
    const int n0 = t << 4;
    const int rowb = (n0 + l16) * LDH;
    v16h A[QMAX];
    #pragma unroll
    for (int q = 0; q < QMAX; ++q) {
      const int b0 = rowb + q * 32 + half * 8;   // halves 0..7
      const int b1 = b0 + 16;                    // halves 8..15
      const float4* x0p = (const float4*)(in0 + b0);
      const float4* x1p = (const float4*)(in0 + b1);
      float4 xA = x0p[0], xB = x0p[1], xC = x1p[0], xD = x1p[1];
      float xv[16] = {xA.x,xA.y,xA.z,xA.w, xB.x,xB.y,xB.z,xB.w,
                      xC.x,xC.y,xC.z,xC.w, xD.x,xD.y,xD.z,xD.w};
      if (MODE == 0) {
        const float4* y0p = (const float4*)(in1 + b0);
        const float4* y1p = (const float4*)(in1 + b1);
        float4 yA = y0p[0], yB = y0p[1], yC = y1p[0], yD = y1p[1];
        float yv[16] = {yA.x,yA.y,yA.z,yA.w, yB.x,yB.y,yB.z,yB.w,
                        yC.x,yC.y,yC.z,yC.w, yD.x,yD.y,yD.z,yD.w};
        #pragma unroll
        for (int k = 0; k < 16; ++k) xv[k] += yv[k];
      } else {
        #pragma unroll
        for (int k = 0; k < 16; ++k)
          xv[k] = fmaxf(scv[q][k] * xv[k] + biv[q][k], 0.0f);
      }
      v16h av;
      #pragma unroll
      for (int k = 0; k < 16; ++k) av[k] = (_Float16)xv[k];
      A[q] = av;
    }
    #pragma unroll
    for (int c = 0; c < 4; ++c) {
      v8f C;
      #pragma unroll
      for (int r = 0; r < 8; ++r) C[r] = bcol[c];
      #pragma unroll
      for (int q = 0; q < QMAX; ++q)
        C = __builtin_amdgcn_wmma_f32_16x16x32_f16(false, A[q], false, B[c][q],
                                                   (short)0, C, false, false);
      const int col = c * 16 + l16;
      #pragma unroll
      for (int r = 0; r < 8; ++r) {
        float v = C[r];
        out[(n0 + r + 8 * half) * LDH + col] = v;
        psum[c] += v;
        psq[c]  += v * v;
      }
    }
  }
  #pragma unroll
  for (int c = 0; c < 4; ++c) {
    unsafeAtomicAdd(&stats[c * 16 + l16],      psum[c]);
    unsafeAtomicAdd(&stats[64 + c * 16 + l16], psq[c]);
  }
}

// fold BN batch stats into scale/bias (training-mode, biased var)
__global__ void bn_finalize_kernel(const float* __restrict__ stats,
                                   const float* __restrict__ gamma,
                                   const float* __restrict__ beta,
                                   float* __restrict__ sb, float invN)
{
  int ch = threadIdx.x;   // 64 threads
  float mu  = stats[ch] * invN;
  float var = stats[64 + ch] * invN - mu * mu;
  float sc  = gamma[ch] * rsqrtf(var + 1e-5f);
  sb[ch]      = sc;
  sb[64 + ch] = beta[ch] - mu * sc;
}

// h_next = relu(sb*t2 + sb2); write h32 in place; pool per graph (atomic segsum)
__global__ void apply_pool_kernel(const float* __restrict__ t2,
                                  const float* __restrict__ sb,
                                  const int*   __restrict__ batch,
                                  float* __restrict__ h32,
                                  float* __restrict__ pools,   // [G, 192]
                                  int nNodes, int layer)
{
  int i = blockIdx.x * blockDim.x + threadIdx.x;
  if (i >= nNodes * LDH) return;
  int n = i >> 6, ch = i & 63;
  float v = sb[ch] * t2[i] + sb[64 + ch];
  v = fmaxf(v, 0.0f);
  h32[i] = v;
  unsafeAtomicAdd(&pools[batch[n] * 192 + layer * 64 + ch], v);
}

// ---------------------------------------------------------------- launcher

extern "C" void kernel_launch(void* const* d_in, const int* in_sizes, int n_in,
                              void* d_out, int out_size, void* d_ws, size_t ws_size,
                              hipStream_t stream)
{
  const float* x     = (const float*)d_in[0];
  const int*   ei    = (const int*)d_in[1];
  const int*   batch = (const int*)d_in[2];
  const float* eattr = (const float*)d_in[3];
  const int nNodes = in_sizes[0] / 32;
  const int nEdges = in_sizes[1] / 2;
  const int NN = nNodes * LDH;

  // params flattened in dict insertion order:
  // per layer: edge_W, edge_b, [mlp0: W,b,gamma,beta], [mlp1: W,b,gamma,beta]
  const float* P[30];
  for (int i = 0; i < 30 && (4 + i) < n_in; ++i) P[i] = (const float*)d_in[4 + i];

  float* h32   = (float*)d_ws;
  float* aggr  = h32 + NN;
  float* t1    = aggr + NN;
  float* t2    = t1 + NN;
  float* stats = t2 + NN;
  float* sb    = stats + 128;
  float* outf  = (float*)d_out;

  const dim3 blk(256);
  const int zb = (NN + 255) / 256;
  const float invN = 1.0f / (float)nNodes;

  zero_kernel<<<dim3((out_size + 255) / 256), blk, 0, stream>>>(outf, out_size);
  init_h_kernel<<<dim3(zb), blk, 0, stream>>>(x, h32, nNodes);

  for (int l = 0; l < 3; ++l) {
    const float* const* Lp = P + l * 10;

    zero_kernel<<<dim3(zb), blk, 0, stream>>>(aggr, NN);
    if (l == 0)
      edge_scatter_kernel<2><<<dim3(1024), blk, 0, stream>>>(
          eattr, ei, ei + nEdges, h32, Lp[0], Lp[1], aggr, nEdges);
    else
      edge_scatter_kernel<4><<<dim3(1024), blk, 0, stream>>>(
          eattr, ei, ei + nEdges, h32, Lp[0], Lp[1], aggr, nEdges);

    zero_kernel<<<dim3(1), dim3(128), 0, stream>>>(stats, 128);
    if (l == 0)
      linear_wmma_kernel<1, 0><<<dim3(256), blk, 0, stream>>>(
          h32, aggr, (const float*)nullptr, Lp[2], Lp[3], t1, stats, nNodes);
    else
      linear_wmma_kernel<2, 0><<<dim3(256), blk, 0, stream>>>(
          h32, aggr, (const float*)nullptr, Lp[2], Lp[3], t1, stats, nNodes);
    bn_finalize_kernel<<<dim3(1), dim3(64), 0, stream>>>(stats, Lp[4], Lp[5], sb, invN);

    zero_kernel<<<dim3(1), dim3(128), 0, stream>>>(stats, 128);
    linear_wmma_kernel<2, 1><<<dim3(256), blk, 0, stream>>>(
        t1, (const float*)nullptr, sb, Lp[6], Lp[7], t2, stats, nNodes);
    bn_finalize_kernel<<<dim3(1), dim3(64), 0, stream>>>(stats, Lp[8], Lp[9], sb, invN);

    apply_pool_kernel<<<dim3(zb), blk, 0, stream>>>(
        t2, sb, batch, h32, outf, nNodes, l);
  }
  (void)ws_size;
}
